// GAT_87729001988239
// MI455X (gfx1250) — matrix operand order repaired and checked by reading, not compiled, not run on hardware
//
#include <hip/hip_runtime.h>

typedef __attribute__((ext_vector_type(16))) _Float16 v16h;
typedef __attribute__((ext_vector_type(8)))  _Float16 v8h;
typedef __attribute__((ext_vector_type(8)))  float    v8f;

#define NEG_SLOPE 0.2f

#if __has_builtin(__builtin_amdgcn_tensor_load_to_lds) && __has_builtin(__builtin_amdgcn_s_wait_tensorcnt)
#define HAVE_TDM 1
typedef __attribute__((ext_vector_type(4))) unsigned int v4u;
typedef __attribute__((ext_vector_type(8))) int          v8i;
typedef __attribute__((ext_vector_type(4))) int          v4i;
#else
#define HAVE_TDM 0
#endif

// f32 -> f16 conversion pre-pass (one-time per layer; removes cvt chains from GEMM)
__global__ void cvt_f16(const float* __restrict__ in, _Float16* __restrict__ out, int n)
{
  int t = blockIdx.x * blockDim.x + threadIdx.x;
  if (t < n) out[t] = (_Float16)in[t];
}

// ---------------------------------------------------------------------------
// WMMA GEMM: C[N x 128] = A16[N x K] * W16[K x 128]  (f16 in, f32 accum)
// One wave32 per 16x16 tile; 8 waves/block cover the 128 columns; K templated
// so the k-loop fully unrolls into back-to-back v_wmma.
// W16 (K*128*2B <= 32KB) is staged into LDS by the Tensor Data Mover:
//   D# group0: count=1 | lds_addr | global_addr[56:0] | type=2
//   D# group1: data_size=1(2B), tensor_dim0=128, tensor_dim1=K,
//              tile_dim0=128, tile_dim1=K, tensor_dim0_stride=128
// then s_wait_tensorcnt 0 + workgroup barrier.
// Fragment packing per CDNA5 ISA 7.12.2 (wave32):
//   A: lane l -> row M=l%16; a[j]=A[M][k0+8*hf+j], a[8+j]=A[M][k0+16+8*hf+j]
//   B: lane l -> col N=l%16; b[j]=W[k0+8*hf+j][N], b[8+j]=W[k0+16+8*hf+j][N]
//   C: c[r] -> C[row0 + r + 8*hf][col0 + l%16]
// ---------------------------------------------------------------------------
template <int K>
__global__ __launch_bounds__(256) void gemm128_wmma(
    const _Float16* __restrict__ A16, const _Float16* __restrict__ W16,
    float* __restrict__ C)
{
  __shared__ _Float16 sW[K * 128];

#if HAVE_TDM
  if (threadIdx.x < 32) {          // one wave issues the TDM descriptor
    unsigned long long ga = (unsigned long long)W16;
    unsigned int lds = (unsigned int)(unsigned long long)(__SIZE_TYPE__)sW;
    v4u g0 = { 1u,                                   // count=1, user mode
               lds,                                  // lds_addr
               (unsigned int)ga,                     // global_addr[31:0]
               ((unsigned int)(ga >> 32) & 0x01FFFFFFu) | 0x80000000u }; // addr[56:32] | type=2
    v8i g1 = { (int)0x00010000u,                     // data_size=1 (2 bytes)
               (int)(128u << 16),                    // tensor_dim0 = 128
               (int)(((unsigned)K & 0xFFFFu) << 16), // tensor_dim1 = K (low bits)
               (int)(128u << 16),                    // tile_dim0 = 128
               (int)(unsigned)K,                     // tile_dim1 = K
               128,                                  // tensor_dim0_stride = 128
               0, 0 };
    v4i gz4 = { 0, 0, 0, 0 };
    v8i gz8 = { 0, 0, 0, 0, 0, 0, 0, 0 };
    __builtin_amdgcn_tensor_load_to_lds(g0, g1, gz4, gz4, gz8, 0);
    __builtin_amdgcn_s_wait_tensorcnt(0);
  }
#else
  {
    const v8h* srcp = (const v8h*)W16;
    v8h* dstl = (v8h*)sW;
    for (int i = threadIdx.x; i < K * 16; i += 256) dstl[i] = srcp[i];
  }
#endif
  __syncthreads();

  const int wave = threadIdx.x >> 5;   // column tile 0..7
  const int lane = threadIdx.x & 31;
  const int hf   = lane >> 4;
  const int ln   = lane & 15;
  const int row0 = blockIdx.x << 4;
  const int col0 = wave << 4;

  v8f acc = {};
  const _Float16* __restrict__ arow = A16 + (size_t)(row0 + ln) * K;
#pragma unroll
  for (int k0 = 0; k0 < K; k0 += 32) {
    v8h a_lo = *(const v8h*)(arow + k0 + 8 * hf);
    v8h a_hi = *(const v8h*)(arow + k0 + 16 + 8 * hf);
    v16h a, b;
#pragma unroll
    for (int j = 0; j < 8; ++j) { a[j] = a_lo[j]; a[8 + j] = a_hi[j]; }
#pragma unroll
    for (int j = 0; j < 16; ++j) b[j] = sW[(k0 + 8 * hf + (j & 7) + 16 * (j >> 3)) * 128 + col0 + ln];
    acc = __builtin_amdgcn_wmma_f32_16x16x32_f16(
        false, a, false, b, (short)0, acc, false, false);
  }
#pragma unroll
  for (int r = 0; r < 8; ++r)
    C[(size_t)(row0 + r + 8 * hf) * 128 + col0 + ln] = acc[r];
}

// Small GEMM for layer 2: C[N x 4] = A[N x K] * W[K x 4]
__global__ void gemm4(const float* __restrict__ A, const float* __restrict__ W,
                      float* __restrict__ C, int N, int K)
{
  int t = blockIdx.x * blockDim.x + threadIdx.x;
  if (t >= N * 4) return;
  int n = t >> 2, c = t & 3;
  const float* a = A + (size_t)n * K;
  float acc = 0.f;
  for (int k = 0; k < K; ++k) acc += a[k] * W[k * 4 + c];
  C[t] = acc;
}

// el[n,h] = <feat[n,h,:], al[h,:]> ; er likewise.  One thread per (n,h).
__global__ void attn_node(const float* __restrict__ feat,
                          const float* __restrict__ al, const float* __restrict__ ar,
                          float* __restrict__ el, float* __restrict__ er,
                          int N, int H, int D)
{
  int t = blockIdx.x * blockDim.x + threadIdx.x;
  if (t >= N * H) return;
  int h = t % H;
  const float* f  = feat + (size_t)t * D;
  const float* pl = al + h * D;
  const float* pr = ar + h * D;
  float sl = 0.f, sr = 0.f;
  for (int d = 0; d < D; ++d) { float v = f[d]; sl += v * pl[d]; sr += v * pr[d]; }
  el[t] = sl;
  er[t] = sr;
}

__global__ void fill_f32(float* __restrict__ p, float v, int n)
{
  int t = blockIdx.x * blockDim.x + threadIdx.x;
  if (t < n) p[t] = v;
}

// float atomic max via signed/unsigned integer compare trick
__device__ inline void atomicMaxF(float* addr, float val)
{
  if (val >= 0.f) atomicMax((int*)addr, __float_as_int(val));
  else            atomicMin((unsigned int*)addr, __float_as_uint(val));
}

// e = leaky_relu(el[src]+er[dst]); segment-max into m[dst]
template <int H>
__global__ void edge_score(const int* __restrict__ src, const int* __restrict__ dst,
                           const float* __restrict__ el, const float* __restrict__ er,
                           float* __restrict__ esc, float* __restrict__ m, int E)
{
  int t = blockIdx.x * blockDim.x + threadIdx.x;
  if (t >= E * H) return;
  int e = t / H, h = t - e * H;
  int is = src[e], id = dst[e];
  float v = el[(size_t)is * H + h] + er[(size_t)id * H + h];
  v = v > 0.f ? v : NEG_SLOPE * v;
  esc[t] = v;
  atomicMaxF(m + (size_t)id * H + h, v);
}

// zero-in-degree safety: m = isfinite(m) ? m : 0
__global__ void fixup_m(float* __restrict__ m, int n)
{
  int t = blockIdx.x * blockDim.x + threadIdx.x;
  if (t >= n) return;
  float v = m[t];
  if (!__builtin_isfinite(v)) m[t] = 0.f;
}

// ex = exp(e - m[dst]); segment-sum into s[dst]
template <int H>
__global__ void edge_exp(const int* __restrict__ dst, const float* __restrict__ m,
                         float* __restrict__ esc, float* __restrict__ s, int E)
{
  int t = blockIdx.x * blockDim.x + threadIdx.x;
  if (t >= E * H) return;
  int e = t / H, h = t - e * H;
  int id = dst[e];
  float v = expf(esc[t] - m[(size_t)id * H + h]);
  esc[t] = v;
  atomicAdd(s + (size_t)id * H + h, v);
}

// rst[dst,h,:] += feat[src,h,:] * (ex / s[dst,h])  — float4 gather, f32 atomics
template <int H, int D>
__global__ void edge_aggregate(const int* __restrict__ src, const int* __restrict__ dst,
                               const float* __restrict__ feat, const float* __restrict__ esc,
                               const float* __restrict__ s, float* __restrict__ rst, int E)
{
  int t = blockIdx.x * blockDim.x + threadIdx.x;
  if (t >= E * H) return;
  int e = t / H, h = t - e * H;
  int is = src[e], id = dst[e];
  const float* f = feat + ((size_t)is * H + h) * D;
  __builtin_prefetch(f, 0, 1);                       // global_prefetch_b8
  float alpha = esc[t] / s[(size_t)id * H + h];
  float* o = rst + ((size_t)id * H + h) * D;
  const float4* f4 = (const float4*)f;
#pragma unroll
  for (int q = 0; q < (D >> 2); ++q) {
    float4 v = f4[q];
    atomicAdd(o + 4 * q + 0, v.x * alpha);
    atomicAdd(o + 4 * q + 1, v.y * alpha);
    atomicAdd(o + 4 * q + 2, v.z * alpha);
    atomicAdd(o + 4 * q + 3, v.w * alpha);
  }
}

// rst += residual + bias ; optional ELU (alpha=1)
__global__ void finalize(float* __restrict__ rst, const float* __restrict__ resid,
                         const float* __restrict__ bias, int total, int C, int use_act)
{
  int t = blockIdx.x * blockDim.x + threadIdx.x;
  if (t >= total) return;
  int c = t % C;
  float v = rst[t] + bias[c];
  if (resid) v += resid[t];
  if (use_act) v = v > 0.f ? v : expm1f(v);
  rst[t] = v;
}

static inline int cdiv(long long a, long long b) { return (int)((a + b - 1) / b); }

extern "C" void kernel_launch(void* const* d_in, const int* in_sizes, int n_in,
                              void* d_out, int out_size, void* d_ws, size_t ws_size,
                              hipStream_t stream)
{
  const float* x   = (const float*)d_in[0];
  const int*   src = (const int*)d_in[1];
  const int*   dst = (const int*)d_in[2];
  const float* W0  = (const float*)d_in[3];
  const float* al0 = (const float*)d_in[4];
  const float* ar0 = (const float*)d_in[5];
  const float* b0  = (const float*)d_in[6];
  const float* W1  = (const float*)d_in[7];
  const float* al1 = (const float*)d_in[8];
  const float* ar1 = (const float*)d_in[9];
  const float* b1  = (const float*)d_in[10];
  const float* W2  = (const float*)d_in[11];
  const float* al2 = (const float*)d_in[12];
  const float* ar2 = (const float*)d_in[13];
  const float* b2  = (const float*)d_in[14];

  const int IN = 64, HD = 4, HID = 32, F = HD * HID /*128*/, NC = 4;
  const int N = in_sizes[0] / IN;   // 50000
  const int E = in_sizes[1];        // 800000
  (void)n_in; (void)ws_size; (void)out_size;

  float* ws  = (float*)d_ws;
  float* Fb  = ws; ws += (size_t)N * F;   // feat of current layer
  float* H1  = ws; ws += (size_t)N * F;   // layer-0 output
  float* H2  = ws; ws += (size_t)N * F;   // layer-1 output
  float* el  = ws; ws += (size_t)N * HD;
  float* er  = ws; ws += (size_t)N * HD;
  float* mb  = ws; ws += (size_t)N * HD;
  float* sb  = ws; ws += (size_t)N * HD;
  float* esc = ws; ws += (size_t)E * HD;
  _Float16* A16 = (_Float16*)ws; ws += (size_t)N * F / 2;   // N*128 f16
  _Float16* W16 = (_Float16*)ws; ws += (size_t)F * F / 2;   // 128*128 f16

  const int TB = 256;
  const float NEGINF = -__builtin_huge_valf();
  float* out = (float*)d_out;

  // ---------------- Layer 0: 64 -> 4x32, ELU, no residual ----------------
  cvt_f16<<<cdiv((long long)N * IN, TB), TB, 0, stream>>>(x, A16, N * IN);
  cvt_f16<<<cdiv(IN * F, TB), TB, 0, stream>>>(W0, W16, IN * F);
  gemm128_wmma<64><<<N / 16, 256, 0, stream>>>(A16, W16, Fb);
  attn_node<<<cdiv((long long)N * HD, TB), TB, 0, stream>>>(Fb, al0, ar0, el, er, N, HD, HID);
  fill_f32<<<cdiv(N * HD, TB), TB, 0, stream>>>(mb, NEGINF, N * HD);
  (void)hipMemsetAsync(sb, 0, (size_t)N * HD * sizeof(float), stream);
  edge_score<4><<<cdiv((long long)E * HD, TB), TB, 0, stream>>>(src, dst, el, er, esc, mb, E);
  fixup_m<<<cdiv(N * HD, TB), TB, 0, stream>>>(mb, N * HD);
  edge_exp<4><<<cdiv((long long)E * HD, TB), TB, 0, stream>>>(dst, mb, esc, sb, E);
  (void)hipMemsetAsync(H1, 0, (size_t)N * F * sizeof(float), stream);
  edge_aggregate<4, 32><<<cdiv((long long)E * HD, TB), TB, 0, stream>>>(src, dst, Fb, esc, sb, H1, E);
  finalize<<<cdiv((long long)N * F, TB), TB, 0, stream>>>(H1, nullptr, b0, N * F, F, 1);

  // ---------------- Layer 1: 128 -> 4x32, ELU, identity residual ----------
  cvt_f16<<<cdiv((long long)N * F, TB), TB, 0, stream>>>(H1, A16, N * F);
  cvt_f16<<<cdiv(F * F, TB), TB, 0, stream>>>(W1, W16, F * F);
  gemm128_wmma<128><<<N / 16, 256, 0, stream>>>(A16, W16, Fb);
  attn_node<<<cdiv((long long)N * HD, TB), TB, 0, stream>>>(Fb, al1, ar1, el, er, N, HD, HID);
  fill_f32<<<cdiv(N * HD, TB), TB, 0, stream>>>(mb, NEGINF, N * HD);
  (void)hipMemsetAsync(sb, 0, (size_t)N * HD * sizeof(float), stream);
  edge_score<4><<<cdiv((long long)E * HD, TB), TB, 0, stream>>>(src, dst, el, er, esc, mb, E);
  fixup_m<<<cdiv(N * HD, TB), TB, 0, stream>>>(mb, N * HD);
  edge_exp<4><<<cdiv((long long)E * HD, TB), TB, 0, stream>>>(dst, mb, esc, sb, E);
  (void)hipMemsetAsync(H2, 0, (size_t)N * F * sizeof(float), stream);
  edge_aggregate<4, 32><<<cdiv((long long)E * HD, TB), TB, 0, stream>>>(src, dst, Fb, esc, sb, H2, E);
  finalize<<<cdiv((long long)N * F, TB), TB, 0, stream>>>(H2, H1, b1, N * F, F, 1);

  // ---------------- Layer 2: 128 -> 1x4, no act, no residual --------------
  gemm4<<<cdiv((long long)N * NC, TB), TB, 0, stream>>>(H2, W2, Fb, N, F);  // feat2 reuses Fb
  attn_node<<<cdiv(N, TB), TB, 0, stream>>>(Fb, al2, ar2, el, er, N, 1, NC);
  fill_f32<<<cdiv(N, TB), TB, 0, stream>>>(mb, NEGINF, N);
  (void)hipMemsetAsync(sb, 0, (size_t)N * sizeof(float), stream);
  edge_score<1><<<cdiv(E, TB), TB, 0, stream>>>(src, dst, el, er, esc, mb, E);
  fixup_m<<<cdiv(N, TB), TB, 0, stream>>>(mb, N);
  edge_exp<1><<<cdiv(E, TB), TB, 0, stream>>>(dst, mb, esc, sb, E);
  (void)hipMemsetAsync(out, 0, (size_t)N * NC * sizeof(float), stream);
  edge_aggregate<1, 4><<<cdiv(E, TB), TB, 0, stream>>>(src, dst, Fb, esc, sb, out, E);
  // H==1 -> mean over heads is identity; just add bias.
  finalize<<<cdiv((long long)N * NC, TB), TB, 0, stream>>>(out, nullptr, b2, N * NC, NC, 0);
}
